// GABlock_61942018343642
// MI455X (gfx1250) — matrix operand order repaired and checked by reading, not compiled, not run on hardware
//
#include <hip/hip_runtime.h>
#include <hip/hip_bf16.h>

// ---------------------------------------------------------------------------
// GABlock (IPA-style geometric attention) for MI455X / gfx1250.
// All K>=24 contractions run on v_wmma_f32_16x16x32_bf16 (bf16 in, f32 acc).
// Wave32: each wave owns one 16x16 C tile. All fragment loads are
// unconditional (operands pre-padded) and all non-unit strides are
// compile-time constants so strided fragments lower to one s_clause of
// immediate-offset loads. Per-head softplus coefficients are precomputed.
// ---------------------------------------------------------------------------

typedef __attribute__((ext_vector_type(16))) __bf16 v16bf;
typedef __attribute__((ext_vector_type(8)))  float  v8f;

#define LQ    512
#define HN    12
#define DPAIR 64

static __device__ inline v8f wmma_bf16(v16bf a, v16bf b, v8f c) {
  return __builtin_amdgcn_wmma_f32_16x16x32_bf16(false, a, false, b,
                                                 (short)0, c, false, false);
}

// A fragment 16x32 bf16, contiguous k. Lanes 0-15: M=lane, K=0..15;
// lanes 16-31: K=16..31. 4x float4 loads per lane.
static __device__ inline v16bf frag_a_fast(const float* __restrict__ A, long lda,
                                           int m0, int k0, int lane) {
  const float* p = A + (long)(m0 + (lane & 15)) * lda + k0 + ((lane >> 4) << 4);
  float4 x0 = *(const float4*)(p);
  float4 x1 = *(const float4*)(p + 4);
  float4 y0 = *(const float4*)(p + 8);
  float4 y1 = *(const float4*)(p + 12);
  v16bf a;
  a[0] = (__bf16)x0.x; a[1] = (__bf16)x0.y; a[2]  = (__bf16)x0.z; a[3]  = (__bf16)x0.w;
  a[4] = (__bf16)x1.x; a[5] = (__bf16)x1.y; a[6]  = (__bf16)x1.z; a[7]  = (__bf16)x1.w;
  a[8] = (__bf16)y0.x; a[9] = (__bf16)y0.y; a[10] = (__bf16)y0.z; a[11] = (__bf16)y0.w;
  a[12] = (__bf16)y1.x; a[13] = (__bf16)y1.y; a[14] = (__bf16)y1.z; a[15] = (__bf16)y1.w;
  return a;
}

// A fragment with compile-time strides (used for alpha).
template <long RS, int CS>
static __device__ inline v16bf frag_a_strided(const float* __restrict__ A,
                                              int m0, int k0, int lane) {
  const float* p = A + (long)(m0 + (lane & 15)) * RS
                     + (long)(k0 + ((lane >> 4) << 4)) * CS;
  v16bf a;
#pragma unroll
  for (int i = 0; i < 16; ++i) {
    int kk = i;
    a[i] = (__bf16)p[kk * CS];
  }
  return a;
}

// B fragment from row-major [K x LDB]: lanes 0-15 -> N=n0+lane, K=k0..k0+15;
// lanes 16-31 -> K=k0+16..k0+31. 16 immediate-offset b32 loads.
template <int LDB>
static __device__ inline v16bf frag_b_fast(const float* __restrict__ B,
                                           int k0, int n0, int lane) {
  const float* p = B + (long)(k0 + ((lane >> 4) << 4)) * LDB + n0 + (lane & 15);
  v16bf b;
#pragma unroll
  for (int i = 0; i < 16; ++i) b[i] = (__bf16)p[i * LDB];
  return b;
}

// B fragment from row-major [N x K] (multiply by transpose); contiguous k.
static __device__ inline v16bf frag_bt_fast(const float* __restrict__ Bt, long ldbt,
                                            int n0, int k0, int lane) {
  const float* p = Bt + (long)(n0 + (lane & 15)) * ldbt + k0 + ((lane >> 4) << 4);
  float4 x0 = *(const float4*)(p);
  float4 x1 = *(const float4*)(p + 4);
  float4 x2 = *(const float4*)(p + 8);
  float4 x3 = *(const float4*)(p + 12);
  v16bf b;
  b[0] = (__bf16)x0.x; b[1] = (__bf16)x0.y; b[2]  = (__bf16)x0.z; b[3]  = (__bf16)x0.w;
  b[4] = (__bf16)x1.x; b[5] = (__bf16)x1.y; b[6]  = (__bf16)x1.z; b[7]  = (__bf16)x1.w;
  b[8] = (__bf16)x2.x; b[9] = (__bf16)x2.y; b[10] = (__bf16)x2.z; b[11] = (__bf16)x2.w;
  b[12] = (__bf16)x3.x; b[13] = (__bf16)x3.y; b[14] = (__bf16)x3.z; b[15] = (__bf16)x3.w;
  return b;
}

// ---------------------------------------------------------------------------
// Generic WMMA GEMM: C[M,N] = A[M,K] @ B[K,N] (+bias, optional ReLU)
// Preconditions: M%16==0, N%16==0, K%32==0, B readable for full tiles.
// block = 128 threads = 4 waves; wave w -> N-tile (blockIdx.y*4 + w)
// ---------------------------------------------------------------------------
template <int LDB>
__global__ void gemm_kernel(const float* __restrict__ A, int lda,
                            const float* __restrict__ B,
                            float* __restrict__ C, int ldc,
                            const float* __restrict__ bias,
                            int N, int K, int relu) {
  int lane = threadIdx.x & 31;
  int wid  = threadIdx.x >> 5;
  int m0 = blockIdx.x * 16;
  int n0 = (blockIdx.y * 4 + wid) * 16;
  if (n0 >= N) return;
  v8f acc = {};
#pragma unroll 2
  for (int k0 = 0; k0 < K; k0 += 32) {
    __builtin_prefetch(A + (long)(m0 + (lane & 15)) * lda + k0 + 64, 0, 1);
    v16bf a = frag_a_fast(A, lda, m0, k0, lane);
    v16bf b = frag_b_fast<LDB>(B, k0, n0, lane);
    acc = wmma_bf16(a, b, acc);
  }
  int n  = n0 + (lane & 15);
  int mb = m0 + ((lane >> 4) << 3);
  float bv = bias ? bias[n] : 0.0f;           // one load, shared by all 8 rows
  float lo = relu ? 0.0f : -3.0e38f;          // branchless ReLU select
#pragma unroll
  for (int r = 0; r < 8; ++r) {
    float v = fmaxf(acc[r] + bv, lo);
    C[(long)(mb + r) * ldc + n] = v;
  }
}

// Zero-padded copy: dst[KP x NP] <- src[K x N]
__global__ void pad2d_kernel(const float* __restrict__ src, int K, int N,
                             float* __restrict__ dst, int KP, int NP) {
  int i = blockIdx.x * blockDim.x + threadIdx.x;
  if (i >= KP * NP) return;
  int k = i / NP, n = i % NP;
  dst[i] = (k < K && n < N) ? src[k * N + n] : 0.0f;
}

// Precompute per-head spatial coefficient: sh[h] = -softplus(coef[h])/12
__global__ void coef_kernel(const float* __restrict__ spatial_coef,
                            float* __restrict__ sh) {
  int h = threadIdx.x;
  if (h < HN) sh[h] = -log1pf(__expf(spatial_coef[h])) * (1.0f / 12.0f);
}

// ---------------------------------------------------------------------------
// Atom/vec pooling -> afc [L,64] = [mean(atom,32) | mean(vec)@w_vec2s+b | 0pad]
// ---------------------------------------------------------------------------
__global__ void pool_kernel(const float* __restrict__ atomf,
                            const float* __restrict__ vec,
                            const unsigned char* __restrict__ mask_atoms,
                            const float* __restrict__ w_vec2s,
                            const float* __restrict__ b_vec2s,
                            float* __restrict__ afc) {
  int l = blockIdx.x, t = threadIdx.x;
  __shared__ float rv[3];
  __shared__ float cntS;
  if (t == 0) {
    float c = 0.f;
    for (int a = 0; a < 14; ++a) c += (float)mask_atoms[l * 14 + a];
    cntS = c + 1e-8f;
  }
  __syncthreads();
  float cnt = cntS;
  if (t < 32) {
    float s = 0.f;
    for (int a = 0; a < 14; ++a)
      if (mask_atoms[l * 14 + a]) s += atomf[((long)l * 14 + a) * 32 + t];
    afc[(long)l * 64 + t] = s / cnt;
  } else if (t < 35) {
    int c = t - 32;
    float s = 0.f;
    for (int a = 0; a < 14; ++a)
      if (mask_atoms[l * 14 + a]) s += vec[((long)l * 14 + a) * 3 + c];
    rv[c] = s / cnt;
  }
  if (t >= 32 && t < 48) afc[(long)l * 64 + 16 + t] = 0.0f;  // cols 48..63
  __syncthreads();
  if (t >= 48 && t < 64) {
    int j = t - 48;
    float s = b_vec2s[j];
    for (int d = 0; d < 3; ++d) s += rv[d] * w_vec2s[d * 16 + j];
    afc[(long)l * 64 + 32 + j] = s;
  }
}

// local_to_global (q = R p + t) from packed [L, H*24] into padded [L, H*32]
// (columns 24..31 per head zeroed). One thread per (h,p) 3-vector.
__global__ void frames_kernel(const float* __restrict__ R, const float* __restrict__ t,
                              const float* __restrict__ qp, const float* __restrict__ kp,
                              const float* __restrict__ vp,
                              float* __restrict__ qg, float* __restrict__ kg,
                              float* __restrict__ vg) {
  int l = blockIdx.x, j = threadIdx.x; // 0..95
  int h = j >> 3, p = j & 7;
  float Rm[9];
#pragma unroll
  for (int i = 0; i < 9; ++i) Rm[i] = R[l * 9 + i];
  float tx = t[l * 3 + 0], ty = t[l * 3 + 1], tz = t[l * 3 + 2];
  const float* src[3] = {qp, kp, vp};
  float* dst[3] = {qg, kg, vg};
#pragma unroll
  for (int b = 0; b < 3; ++b) {
    const float* s = src[b] + (long)l * 288 + h * 24 + p * 3;
    float x = s[0], y = s[1], z = s[2];
    float* d = dst[b] + (long)l * 384 + h * 32 + p * 3;
    d[0] = Rm[0] * x + Rm[1] * y + Rm[2] * z + tx;
    d[1] = Rm[3] * x + Rm[4] * y + Rm[5] * z + ty;
    d[2] = Rm[6] * x + Rm[7] * y + Rm[8] * z + tz;
    dst[b][(long)l * 384 + h * 32 + 24 + p] = 0.0f;   // zero pad col
  }
}

__global__ void sq_kernel(const float* __restrict__ qg, const float* __restrict__ kg,
                          float* __restrict__ qg2, float* __restrict__ kg2) {
  int l = blockIdx.x, h = threadIdx.x;
  if (h < HN) {
    float s = 0.f, s2 = 0.f;
    for (int c = 0; c < 24; ++c) {
      float a = qg[(long)l * 384 + h * 32 + c]; s  += a * a;
      float b = kg[(long)l * 384 + h * 32 + c]; s2 += b * b;
    }
    qg2[l * HN + h] = s;
    kg2[l * HN + h] = s2;
  }
}

// ---------------------------------------------------------------------------
// logits[q,k,h] = 0.5*(node + atom + spatial) - mask penalty (pair added later)
// node+atom chained through one WMMA accumulator; spatial dot uses the padded
// stride-32 point buffers (pad columns are zero -> exact K=24 dot).
// grid (32,32,3), block 128 -> head = z*4 + wave
// ---------------------------------------------------------------------------
__global__ void logits_kernel(const float* __restrict__ ql, const float* __restrict__ kl,
                              const float* __restrict__ qa, const float* __restrict__ ka,
                              const float* __restrict__ qg, const float* __restrict__ kg,
                              const float* __restrict__ qg2, const float* __restrict__ kg2,
                              const float* __restrict__ shArr,
                              const unsigned char* __restrict__ mask,
                              float* __restrict__ logits) {
  int lane = threadIdx.x & 31;
  int h = blockIdx.z * 4 + (threadIdx.x >> 5);
  int q0 = blockIdx.x * 16, k0 = blockIdx.y * 16;
  v8f acc = {};
  {
    v16bf a = frag_a_fast(ql + h * 32, 384, q0, 0, lane);
    v16bf b = frag_bt_fast(kl + h * 32, 384, k0, 0, lane);
    acc = wmma_bf16(a, b, acc);
    a = frag_a_fast(qa + h * 32, 384, q0, 0, lane);
    b = frag_bt_fast(ka + h * 32, 384, k0, 0, lane);
    acc = wmma_bf16(a, b, acc);
  }
  v8f accs = {};
  {
    v16bf a = frag_a_fast(qg + h * 32, 384, q0, 0, lane);
    v16bf b = frag_bt_fast(kg + h * 32, 384, k0, 0, lane);
    accs = wmma_bf16(a, b, accs);
  }
  float sh = shArr[h];                         // precomputed -softplus/12
  const float rsq = 0.17677669529663687f;      // 1/sqrt(QK)
  int n  = k0 + (lane & 15);
  int mb = q0 + ((lane >> 4) << 3);
  float kg2n = kg2[n * HN + h];
  float mk = mask[n] ? 0.0f : -100000.0f;
#pragma unroll
  for (int r = 0; r < 8; ++r) {
    int m = mb + r;
    float lg = rsq * acc[r]
             + sh * (qg2[m * HN + h] + kg2n)
             - 2.0f * sh * accs[r];
    lg = lg * 0.5f + mk + (mask[m] ? 0.0f : -100000.0f);
    logits[((long)m * LQ + n) * HN + h] = lg;
  }
}

// logits += 0.5 * z[q] @ w_pb_pad    (per q: [512,64]@[64,16], store n<12)
__global__ void pair_logits_kernel(const float* __restrict__ z,
                                   const float* __restrict__ w_pb,
                                   float* __restrict__ logits) {
  int lane = threadIdx.x & 31;
  int wid  = threadIdx.x >> 5;
  int q  = blockIdx.x;
  int m0 = (blockIdx.y * 4 + wid) * 16;
  const float* A = z + (long)q * LQ * DPAIR;
  v8f acc = {};
#pragma unroll
  for (int k0 = 0; k0 < DPAIR; k0 += 32) {
    v16bf a = frag_a_fast(A, DPAIR, m0, k0, lane);
    v16bf b = frag_b_fast<16>(w_pb, k0, 0, lane);
    acc = wmma_bf16(a, b, acc);
  }
  int n  = lane & 15;
  int mb = m0 + ((lane >> 4) << 3);
  if (n < HN) {
#pragma unroll
    for (int r = 0; r < 8; ++r)
      logits[((long)q * LQ + mb + r) * HN + n] += 0.5f * acc[r];
  }
}

// softmax over k (stride HN), in place; zero rows where !mask[q]
__global__ void softmax_kernel(float* __restrict__ logits,
                               const unsigned char* __restrict__ mask) {
  int q = blockIdx.x / HN, h = blockIdx.x % HN;
  int tid = threadIdx.x;
  __shared__ float red[8];
  float* row = logits + (long)q * LQ * HN + h;
  float vmax = -1e30f;
  for (int k = tid; k < LQ; k += 256) vmax = fmaxf(vmax, row[(long)k * HN]);
  for (int off = 16; off; off >>= 1) vmax = fmaxf(vmax, __shfl_xor(vmax, off, 32));
  if ((tid & 31) == 0) red[tid >> 5] = vmax;
  __syncthreads();
  float bmax = red[0];
  for (int i = 1; i < 8; ++i) bmax = fmaxf(bmax, red[i]);
  __syncthreads();
  float s = 0.f;
  for (int k = tid; k < LQ; k += 256) s += __expf(row[(long)k * HN] - bmax);
  for (int off = 16; off; off >>= 1) s += __shfl_xor(s, off, 32);
  if ((tid & 31) == 0) red[tid >> 5] = s;
  __syncthreads();
  float bsum = 0.f;
  for (int i = 0; i < 8; ++i) bsum += red[i];
  float inv = (mask[q] ? 1.0f : 0.0f) / bsum;
  for (int k = tid; k < LQ; k += 256)
    row[(long)k * HN] = __expf(row[(long)k * HN] - bmax) * inv;
}

// C[q, cOff + h*hsC + n] = sum_k alpha[q,k,h] * B[k, h*32 + n]
// grid (8, ceil(Nh/16), 12), block 128. B is a padded stride-384 buffer.
__global__ void attn_gemm_kernel(const float* __restrict__ alpha,
                                 const float* __restrict__ Bbase,
                                 float* __restrict__ C, int ldc, int cOff, int hsC,
                                 int Nh) {
  int lane = threadIdx.x & 31;
  int wid  = threadIdx.x >> 5;
  int h  = blockIdx.z;
  int m0 = (blockIdx.x * 4 + wid) * 16;
  int n0 = blockIdx.y * 16;
  const float* A = alpha + h;                     // A[m][k] = alpha[(m*L+k)*H + h]
  const float* B = Bbase + h * 32;
  v8f acc = {};
#pragma unroll 2
  for (int k0 = 0; k0 < LQ; k0 += 32) {
    v16bf a = frag_a_strided<(long)LQ * HN, HN>(A, m0, k0, lane);
    v16bf b = frag_b_fast<384>(B, k0, n0, lane);
    acc = wmma_bf16(a, b, acc);
  }
  int n  = n0 + (lane & 15);
  int mb = m0 + ((lane >> 4) << 3);
  if (n < Nh) {
#pragma unroll
    for (int r = 0; r < 8; ++r)
      C[(long)(mb + r) * ldc + cOff + h * hsC + n] = acc[r];
  }
}

// featAll[q, h*64+c] = sum_k alpha[q,k,h] * z[q,k,c]   (M=12 padded to 16)
__global__ void p2n_kernel(const float* __restrict__ alpha,
                           const float* __restrict__ z,
                           float* __restrict__ featAll) {
  int lane = threadIdx.x & 31;
  int wid  = threadIdx.x >> 5;  // n-tile 0..3
  int q  = blockIdx.x;
  int n0 = wid * 16;
  const float* A = alpha + (long)q * LQ * HN;      // A[m][k] = alpha[..+k*H+m]
  const float* B = z + (long)q * LQ * DPAIR;
  v8f acc = {};
#pragma unroll 2
  for (int k0 = 0; k0 < LQ; k0 += 32) {
    v16bf a = frag_a_strided<1, HN>(A, 0, k0, lane);
    v16bf b = frag_b_fast<DPAIR>(B, k0, n0, lane);
    acc = wmma_bf16(a, b, acc);
  }
  int n  = n0 + (lane & 15);
  int mb = (lane >> 4) << 3;
#pragma unroll
  for (int r = 0; r < 8; ++r) {
    int m = mb + r;
    if (m < HN) featAll[(long)q * 2208 + m * 64 + n] = acc[r];
  }
}

// global_to_local + norm + dir into featAll[1152:1824]
__global__ void spatial_feat_kernel(const float* __restrict__ R, const float* __restrict__ t,
                                    const float* __restrict__ aggr,
                                    float* __restrict__ featAll) {
  int l = blockIdx.x, j = threadIdx.x;  // j = h*8+p
  int h = j >> 3, p = j & 7;
  const float* g = aggr + (long)l * 288 + h * 24 + p * 3;
  float qx = g[0] - t[l * 3 + 0];
  float qy = g[1] - t[l * 3 + 1];
  float qz = g[2] - t[l * 3 + 2];
  const float* Rm = R + l * 9;
  float fx = Rm[0] * qx + Rm[3] * qy + Rm[6] * qz;   // R^T q
  float fy = Rm[1] * qx + Rm[4] * qy + Rm[7] * qz;
  float fz = Rm[2] * qx + Rm[5] * qy + Rm[8] * qz;
  float fd = sqrtf(fx * fx + fy * fy + fz * fz);
  float inv = 1.0f / fmaxf(fd, 1e-4f);
  float* F = featAll + (long)l * 2208;
  F[1152 + j * 3 + 0] = fx; F[1152 + j * 3 + 1] = fy; F[1152 + j * 3 + 2] = fz;
  F[1440 + j] = fd;
  F[1536 + j * 3 + 0] = fx * inv; F[1536 + j * 3 + 1] = fy * inv; F[1536 + j * 3 + 2] = fz * inv;
}

// out = LN(a + (masked) b) * g + bb   (row of 256, block 256)
__global__ void ln_residual_kernel(const float* __restrict__ a, const float* __restrict__ b,
                                   const unsigned char* __restrict__ mask, int useMask,
                                   const float* __restrict__ g, const float* __restrict__ bb,
                                   float* __restrict__ out) {
  int l = blockIdx.x, c = threadIdx.x;
  __shared__ float red[8];
  float bv = b[(long)l * 256 + c];
  if (useMask && !mask[l]) bv = 0.0f;
  float v = a[(long)l * 256 + c] + bv;
  float s = v;
  for (int off = 16; off; off >>= 1) s += __shfl_xor(s, off, 32);
  if ((c & 31) == 0) red[c >> 5] = s;
  __syncthreads();
  float mu = 0.f;
  for (int i = 0; i < 8; ++i) mu += red[i];
  mu *= (1.0f / 256.0f);
  __syncthreads();
  float d = v - mu;
  float s2 = d * d;
  for (int off = 16; off; off >>= 1) s2 += __shfl_xor(s2, off, 32);
  if ((c & 31) == 0) red[c >> 5] = s2;
  __syncthreads();
  float var = 0.f;
  for (int i = 0; i < 8; ++i) var += red[i];
  var *= (1.0f / 256.0f);
  out[(long)l * 256 + c] = d * rsqrtf(var + 1e-5f) * g[c] + bb[c];
}

// ---------------------------------------------------------------------------
extern "C" void kernel_launch(void* const* d_in, const int* in_sizes, int n_in,
                              void* d_out, int out_size, void* d_ws, size_t ws_size,
                              hipStream_t stream) {
  const float* R        = (const float*)d_in[0];
  const float* t        = (const float*)d_in[1];
  const float* x        = (const float*)d_in[2];
  const float* z        = (const float*)d_in[3];
  const float* aaa_feat = (const float*)d_in[4];
  const float* vec      = (const float*)d_in[5];
  const unsigned char* mask       = (const unsigned char*)d_in[6];
  const unsigned char* mask_atoms = (const unsigned char*)d_in[7];
  const float* w_vec2s  = (const float*)d_in[8];
  const float* b_vec2s  = (const float*)d_in[9];
  const float* w_aq = (const float*)d_in[10]; const float* b_aq = (const float*)d_in[11];
  const float* w_ak = (const float*)d_in[12]; const float* b_ak = (const float*)d_in[13];
  const float* w_av = (const float*)d_in[14]; const float* b_av = (const float*)d_in[15];
  const float* w_q  = (const float*)d_in[16];
  const float* w_k  = (const float*)d_in[17];
  const float* w_v  = (const float*)d_in[18];
  const float* w_pb = (const float*)d_in[19];
  const float* spatial_coef = (const float*)d_in[20];
  const float* w_qp = (const float*)d_in[21];
  const float* w_kp = (const float*)d_in[22];
  const float* w_vp = (const float*)d_in[23];
  const float* w_out = (const float*)d_in[24]; const float* b_out = (const float*)d_in[25];
  const float* ln1_g = (const float*)d_in[26]; const float* ln1_b = (const float*)d_in[27];
  const float* ln2_g = (const float*)d_in[28]; const float* ln2_b = (const float*)d_in[29];
  const float* w_m1 = (const float*)d_in[30]; const float* b_m1 = (const float*)d_in[31];
  const float* w_m2 = (const float*)d_in[32]; const float* b_m2 = (const float*)d_in[33];
  const float* w_m3 = (const float*)d_in[34]; const float* b_m3 = (const float*)d_in[35];

  float* W = (float*)d_ws;
  float* afc   = W; W += 512 * 64;                  // padded K 48->64
  float* waqP  = W; W += 64 * 384;                  // padded weights
  float* wakP  = W; W += 64 * 384;
  float* wavP  = W; W += 64 * 384;
  float* wpbP  = W; W += 64 * 16;                   // padded N 12->16
  float* shArr = W; W += 16;                        // precomputed -softplus/12
  float* qa    = W; W += 512 * 384;
  float* ka    = W; W += 512 * 384;
  float* va    = W; W += 512 * 384;
  float* ql    = W; W += 512 * 384;
  float* kl    = W; W += 512 * 384;
  float* vl    = W; W += 512 * 384;
  float* qp    = W; W += 512 * 288;                 // local-frame projections
  float* kp    = W; W += 512 * 288;
  float* vp    = W; W += 512 * 288;
  float* qg    = W; W += 512 * 384;                 // global points, stride-32/head
  float* kg    = W; W += 512 * 384;
  float* vg    = W; W += 512 * 384;
  float* qg2   = W; W += 512 * 12;
  float* kg2   = W; W += 512 * 12;
  float* logit = W; W += (long)512 * 512 * 12;      // becomes alpha after softmax
  float* aggr  = W; W += 512 * 288;
  float* featA = W; W += 512 * 2208;
  float* outf  = W; W += 512 * 256;
  float* x1    = W; W += 512 * 256;
  float* t1    = W; W += 512 * 256;
  float* t2    = W; W += 512 * 256;
  (void)ws_size; (void)n_in; (void)in_sizes; (void)out_size;

  // 0) pad weights + precompute softplus coefficients
  pad2d_kernel<<<96, 256, 0, stream>>>(w_aq, 48, 384, waqP, 64, 384);
  pad2d_kernel<<<96, 256, 0, stream>>>(w_ak, 48, 384, wakP, 64, 384);
  pad2d_kernel<<<96, 256, 0, stream>>>(w_av, 48, 384, wavP, 64, 384);
  pad2d_kernel<<<4, 256, 0, stream>>>(w_pb, 64, 12, wpbP, 64, 16);
  coef_kernel<<<1, 32, 0, stream>>>(spatial_coef, shArr);

  // 1) atom/vec pooling -> afc (64-wide, zero-padded)
  pool_kernel<<<512, 64, 0, stream>>>(aaa_feat, vec, mask_atoms, w_vec2s, b_vec2s, afc);

  // 2) projections (WMMA GEMMs)
  dim3 g384(32, 6), g288(32, 5), g256(32, 4);
  gemm_kernel<384><<<g384, 128, 0, stream>>>(afc, 64, waqP, qa, 384, b_aq, 384, 64, 0);
  gemm_kernel<384><<<g384, 128, 0, stream>>>(afc, 64, wakP, ka, 384, b_ak, 384, 64, 0);
  gemm_kernel<384><<<g384, 128, 0, stream>>>(afc, 64, wavP, va, 384, b_av, 384, 64, 0);
  gemm_kernel<384><<<g384, 128, 0, stream>>>(x, 256, w_q, ql, 384, nullptr, 384, 256, 0);
  gemm_kernel<384><<<g384, 128, 0, stream>>>(x, 256, w_k, kl, 384, nullptr, 384, 256, 0);
  gemm_kernel<384><<<g384, 128, 0, stream>>>(x, 256, w_v, vl, 384, nullptr, 384, 256, 0);
  gemm_kernel<288><<<g288, 128, 0, stream>>>(x, 256, w_qp, qp, 288, nullptr, 288, 256, 0);
  gemm_kernel<288><<<g288, 128, 0, stream>>>(x, 256, w_kp, kp, 288, nullptr, 288, 256, 0);
  gemm_kernel<288><<<g288, 128, 0, stream>>>(x, 256, w_vp, vp, 288, nullptr, 288, 256, 0);

  // 3) frame transforms (into padded stride-32 layout) + squared norms
  frames_kernel<<<512, 96, 0, stream>>>(R, t, qp, kp, vp, qg, kg, vg);
  sq_kernel<<<512, 32, 0, stream>>>(qg, kg, qg2, kg2);

  // 4) logits (node+atom+spatial), pair bias, softmax
  logits_kernel<<<dim3(32, 32, 3), 128, 0, stream>>>(ql, kl, qa, ka, qg, kg,
                                                     qg2, kg2, shArr, mask, logit);
  pair_logits_kernel<<<dim3(512, 8), 128, 0, stream>>>(z, wpbP, logit);
  softmax_kernel<<<512 * 12, 256, 0, stream>>>(logit, mask);

  // 5) alpha-weighted aggregations (write straight into featAll slices)
  attn_gemm_kernel<<<dim3(8, 2, 12), 128, 0, stream>>>(logit, vl,
                                                       featA, 2208, 768, 32, 32);   // feat_node
  attn_gemm_kernel<<<dim3(8, 2, 12), 128, 0, stream>>>(logit, va,
                                                       featA, 2208, 1824, 32, 32);  // feat_atom
  attn_gemm_kernel<<<dim3(8, 2, 12), 128, 0, stream>>>(logit, vg,
                                                       aggr, 288, 0, 24, 24);       // point aggr
  p2n_kernel<<<512, 128, 0, stream>>>(logit, z, featA);                             // feat_p2n
  spatial_feat_kernel<<<512, 96, 0, stream>>>(R, t, aggr, featA);                   // feat_spatial

  // 6) output projection + residual LN + MLP + final LN
  gemm_kernel<256><<<g256, 128, 0, stream>>>(featA, 2208, w_out, outf, 256, b_out,
                                             256, 2208, 0);
  ln_residual_kernel<<<512, 256, 0, stream>>>(x, outf, mask, 1, ln1_g, ln1_b, x1);
  gemm_kernel<256><<<g256, 128, 0, stream>>>(x1, 256, w_m1, t1, 256, b_m1, 256, 256, 1);
  gemm_kernel<256><<<g256, 128, 0, stream>>>(t1, 256, w_m2, t2, 256, b_m2, 256, 256, 1);
  gemm_kernel<256><<<g256, 128, 0, stream>>>(t2, 256, w_m3, t1, 256, b_m3, 256, 256, 0);
  ln_residual_kernel<<<512, 256, 0, stream>>>(x1, t1, mask, 0, ln2_g, ln2_b, (float*)d_out);
}
